// VisionTransformerPredictor_4982162063857
// MI455X (gfx1250) — matrix operand order repaired and checked by reading, not compiled, not run on hardware
//
#include <hip/hip_runtime.h>
#include <hip/hip_bf16.h>

typedef __attribute__((ext_vector_type(16))) _Float16 v16h;
typedef __attribute__((ext_vector_type(8)))  _Float16 v8h;
typedef __attribute__((ext_vector_type(8)))  float    v8f;

#define BB      4
#define SEQ     1536
#define K_ENC   1024
#define K_PRED  512
#define EMBED   768
#define PDIM    384
#define H3      1152
#define HID     1536
#define HEADS   12
#define HD      32
#define DEPTH   6
#define ROWS    (BB * SEQ)          // 6144
#define SCALE_F 0.17677669529663687f
#define LN_EPS  1e-5f

__device__ __forceinline__ v8f wmma16(v16h a, v16h b, v8f c) {
    return __builtin_amdgcn_wmma_f32_16x16x32_f16(false, a, false, b, (short)0, c, false, false);
}

// A-fragment (16x32): per lane, elements are two contiguous 8-half runs:
//   h=0..7  -> K = g*8 + h          h=8..15 -> K = 16 + g*8 + (h-8)
__device__ __forceinline__ v16h load_a_frag(const _Float16* base, int g) {
    v8h lo = *(const v8h*)(base + g * 8);
    v8h hi = *(const v8h*)(base + 16 + g * 8);
    return __builtin_shufflevector(lo, hi, 0, 1, 2, 3, 4, 5, 6, 7,
                                           8, 9, 10, 11, 12, 13, 14, 15);
}

// B-fragment (32x16): per lane (col fixed), h=0..7 -> K = g*8 + h, h=8..15 -> K = 16 + g*8 + (h-8)
__device__ __forceinline__ v16h load_b_frag(const _Float16* base, int g) {
    v8h lo = *(const v8h*)(base + g * 8);
    v8h hi = *(const v8h*)(base + 16 + g * 8);
    return __builtin_shufflevector(lo, hi, 0, 1, 2, 3, 4, 5, 6, 7,
                                           8, 9, 10, 11, 12, 13, 14, 15);
}

// ---------------------------------------------------------------------------
// Weight conversion: src (L,K,N) fp32 -> dst (L,N,K) f16 (transposed per layer)
// ---------------------------------------------------------------------------
__global__ void cvt_transpose(const float* __restrict__ src, _Float16* __restrict__ dst,
                              int K, int N, long total) {
    long i = (long)blockIdx.x * blockDim.x + threadIdx.x;
    if (i >= total) return;
    long kn = (long)K * N;
    long l = i / kn, rem = i - l * kn;
    long k = rem / N, n = rem - k * N;
    dst[l * kn + n * K + k] = (_Float16)src[i];
}

// ---------------------------------------------------------------------------
// Build pre-embed tokens (f16): X0[b,s,:] = x[b,s]+pe[me]  or  mask_token+pe[mp]
// ---------------------------------------------------------------------------
__global__ void build_x0(const float* __restrict__ x, const float* __restrict__ pe,
                         const int* __restrict__ me, const int* __restrict__ mp,
                         const float* __restrict__ mtok, _Float16* __restrict__ X0) {
    int row = blockIdx.x;
    int b = row / SEQ, s = row % SEQ;
    _Float16* o = X0 + (size_t)row * EMBED;
    if (s < K_ENC) {
        int pi = me[b * K_ENC + s];
        const float* xr = x + ((size_t)(b * K_ENC + s)) * EMBED;
        const float* pr = pe + (size_t)pi * EMBED;
        for (int c = threadIdx.x; c < EMBED; c += blockDim.x)
            o[c] = (_Float16)(xr[c] + pr[c]);
    } else {
        int pi = mp[b * K_PRED + (s - K_ENC)];
        const float* pr = pe + (size_t)pi * EMBED;
        for (int c = threadIdx.x; c < EMBED; c += blockDim.x)
            o[c] = (_Float16)(mtok[c] + pr[c]);
    }
}

// ---------------------------------------------------------------------------
// LayerNorm over 384 features (fp32 in, f16 out), one block (128 thr) per row
// ---------------------------------------------------------------------------
__global__ __launch_bounds__(128) void ln_kernel(const float* __restrict__ X,
                                                 const float* __restrict__ w,
                                                 const float* __restrict__ bb,
                                                 _Float16* __restrict__ O) {
    __shared__ float red[128];
    int row = blockIdx.x, tid = threadIdx.x;
    const float* xr = X + (size_t)row * PDIM;
    float v0 = xr[tid], v1 = xr[tid + 128], v2 = xr[tid + 256];
    red[tid] = v0 + v1 + v2;
    for (int st = 64; st > 0; st >>= 1) { __syncthreads(); if (tid < st) red[tid] += red[tid + st]; }
    __syncthreads();
    float mean = red[0] * (1.0f / PDIM);
    __syncthreads();
    float d0 = v0 - mean, d1 = v1 - mean, d2 = v2 - mean;
    red[tid] = d0 * d0 + d1 * d1 + d2 * d2;
    for (int st = 64; st > 0; st >>= 1) { __syncthreads(); if (tid < st) red[tid] += red[tid + st]; }
    __syncthreads();
    float rstd = rsqrtf(red[0] * (1.0f / PDIM) + LN_EPS);
    _Float16* orow = O + (size_t)row * PDIM;
    orow[tid]       = (_Float16)(d0 * rstd * w[tid]       + bb[tid]);
    orow[tid + 128] = (_Float16)(d1 * rstd * w[tid + 128] + bb[tid + 128]);
    orow[tid + 256] = (_Float16)(d2 * rstd * w[tid + 256] + bb[tid + 256]);
}

// ---------------------------------------------------------------------------
// WMMA GEMM:  O = epilogue(A[M,K] @ W[K,N] + bias)
//   A:  f16 row-major (M,K)
//   Wt: f16 TRANSPOSED (N,K)
//   fuse: 0=bias  1=bias+GELU(exact)  2=bias+residual(R fp32)
//   out: fp32 (O32) or f16 (O16)
//   block = 128 thr = 4 waves; wave tile 32(M)x64(N); block tile 128x64
//   Requires M%128==0, N%64==0, K%32==0
// ---------------------------------------------------------------------------
__global__ __launch_bounds__(128) void gemm_wmma_kernel(
    const _Float16* __restrict__ A, const _Float16* __restrict__ Wt,
    const float* __restrict__ bias, const float* __restrict__ R,
    float* __restrict__ O32, _Float16* __restrict__ O16,
    int M, int N, int K, int fuse) {
    const int tid = threadIdx.x;
    const int wave = tid >> 5, lane = tid & 31;
    const int lane16 = lane & 15, g = lane >> 4;
    const int m0 = blockIdx.y * 128 + wave * 32;
    const int n0 = blockIdx.x * 64;

    const _Float16* a0p = A + (size_t)(m0 + lane16) * K;
    const _Float16* a1p = A + (size_t)(m0 + 16 + lane16) * K;
    const _Float16* b0p = Wt + (size_t)(n0 + lane16) * K;
    const _Float16* b1p = Wt + (size_t)(n0 + 16 + lane16) * K;
    const _Float16* b2p = Wt + (size_t)(n0 + 32 + lane16) * K;
    const _Float16* b3p = Wt + (size_t)(n0 + 48 + lane16) * K;

    v8f acc[2][4];
#pragma unroll
    for (int i = 0; i < 2; ++i)
#pragma unroll
        for (int t = 0; t < 4; ++t) acc[i][t] = v8f{};

    for (int kk = 0; kk < K; kk += 32) {
        v16h a0 = load_a_frag(a0p + kk, g);
        v16h a1 = load_a_frag(a1p + kk, g);
        v16h b0 = load_b_frag(b0p + kk, g);
        v16h b1 = load_b_frag(b1p + kk, g);
        v16h b2 = load_b_frag(b2p + kk, g);
        v16h b3 = load_b_frag(b3p + kk, g);
        acc[0][0] = wmma16(a0, b0, acc[0][0]);
        acc[0][1] = wmma16(a0, b1, acc[0][1]);
        acc[0][2] = wmma16(a0, b2, acc[0][2]);
        acc[0][3] = wmma16(a0, b3, acc[0][3]);
        acc[1][0] = wmma16(a1, b0, acc[1][0]);
        acc[1][1] = wmma16(a1, b1, acc[1][1]);
        acc[1][2] = wmma16(a1, b2, acc[1][2]);
        acc[1][3] = wmma16(a1, b3, acc[1][3]);
    }

#pragma unroll
    for (int i = 0; i < 2; ++i) {
#pragma unroll
        for (int t = 0; t < 4; ++t) {
            int n = n0 + t * 16 + lane16;
            float bv = bias[n];
#pragma unroll
            for (int r = 0; r < 8; ++r) {
                int m = m0 + i * 16 + g * 8 + r;
                float v = acc[i][t][r] + bv;
                if (fuse == 1) {
                    v = 0.5f * v * (1.0f + erff(v * 0.70710678118654752f));
                } else if (fuse == 2) {
                    v += R[(size_t)m * N + n];
                }
                size_t idx = (size_t)m * N + n;
                if (O16) O16[idx] = (_Float16)v;
                else     O32[idx] = v;
            }
        }
    }
}

// ---------------------------------------------------------------------------
// Transpose V out of qkvh: vT[((b*HEADS+h)*HD+d), s] = qkvh[(b,s), 768+h*32+d]
// ---------------------------------------------------------------------------
__global__ void transpose_v(const _Float16* __restrict__ qkvh, _Float16* __restrict__ vT) {
    int i = blockIdx.x * blockDim.x + threadIdx.x;   // total BB*HEADS*HD*SEQ
    if (i >= BB * HEADS * HD * SEQ) return;
    int s = i % SEQ;
    int r = i / SEQ;
    int d = r % HD;
    int hh = (r / HD) % HEADS;
    int b = r / (HD * HEADS);
    vT[i] = qkvh[((size_t)(b * SEQ + s)) * H3 + 2 * PDIM + hh * HD + d];
}

// ---------------------------------------------------------------------------
// Attention: one block (128 thr = 4 waves) per (16-query block, head, batch)
// qkvh: (B, SEQ, 1152) f16;  vT: (B*HEADS*HD, SEQ) f16;  out: (B, SEQ, 384) f16
// ---------------------------------------------------------------------------
#define SCLD (SEQ + 8)
__global__ __launch_bounds__(128) void attn_kernel(const _Float16* __restrict__ qkvh,
                                                   const _Float16* __restrict__ vT,
                                                   _Float16* __restrict__ out) {
    __shared__ _Float16 sc[16][SCLD];
    __shared__ float red[16][8];
    __shared__ float outb[4][16][32];

    const int tid = threadIdx.x;
    const int wave = tid >> 5, lane = tid & 31;
    const int lane16 = lane & 15, g = lane >> 4;
    const int q0 = blockIdx.x * 16;
    const int head = blockIdx.y;
    const int b = blockIdx.z;

    // ---- scores: q kT * SCALE (one wmma per 16-key tile, K = HD = 32) ----
    v16h aq = load_a_frag(qkvh + ((size_t)(b * SEQ + q0 + lane16)) * H3 + head * HD, g);
    for (int j = wave; j < SEQ / 16; j += 4) {
        const _Float16* kp = qkvh + ((size_t)(b * SEQ + j * 16 + lane16)) * H3 + PDIM + head * HD;
        v16h bk = load_b_frag(kp, g);
        v8f c = v8f{};
        c = wmma16(aq, bk, c);
#pragma unroll
        for (int r = 0; r < 8; ++r)
            sc[g * 8 + r][j * 16 + lane16] = (_Float16)(c[r] * SCALE_F);
    }
    __syncthreads();

    // ---- softmax (16 rows x 8 threads each) ----
    {
        int row = tid >> 3, sub = tid & 7;
        float lmax = -1e30f;
        for (int c = sub; c < SEQ; c += 8) lmax = fmaxf(lmax, (float)sc[row][c]);
        red[row][sub] = lmax;
        __syncthreads();
        float rmax = red[row][0];
#pragma unroll
        for (int k = 1; k < 8; ++k) rmax = fmaxf(rmax, red[row][k]);
        __syncthreads();
        float lsum = 0.0f;
        for (int c = sub; c < SEQ; c += 8) {
            float e = __expf((float)sc[row][c] - rmax);
            sc[row][c] = (_Float16)e;
            lsum += e;
        }
        red[row][sub] = lsum;
        __syncthreads();
        float rsum = red[row][0];
#pragma unroll
        for (int k = 1; k < 8; ++k) rsum += red[row][k];
        float rinv = 1.0f / rsum;
        __syncthreads();
        for (int c = sub; c < SEQ; c += 8)
            sc[row][c] = (_Float16)((float)sc[row][c] * rinv);
    }
    __syncthreads();

    // ---- probs @ V : each wave handles SEQ/4 keys, K-steps of 32 ----
    {
        const int tbase = wave * (SEQ / 4);
        const _Float16* v0p = vT + ((size_t)((b * HEADS + head) * HD) + lane16) * SEQ;
        const _Float16* v1p = vT + ((size_t)((b * HEADS + head) * HD) + 16 + lane16) * SEQ;
        v8f o0 = v8f{}, o1 = v8f{};
        for (int i = 0; i < (SEQ / 4) / 32; ++i) {
            int kk = tbase + i * 32;
            v16h ap = load_a_frag(&sc[lane16][kk], g);
            v16h bv0 = load_b_frag(v0p + kk, g);
            v16h bv1 = load_b_frag(v1p + kk, g);
            o0 = wmma16(ap, bv0, o0);
            o1 = wmma16(ap, bv1, o1);
        }
#pragma unroll
        for (int r = 0; r < 8; ++r) {
            outb[wave][g * 8 + r][lane16]      = o0[r];
            outb[wave][g * 8 + r][16 + lane16] = o1[r];
        }
    }
    __syncthreads();
    for (int e = tid; e < 512; e += 128) {
        int m = e >> 5, d = e & 31;
        float s = outb[0][m][d] + outb[1][m][d] + outb[2][m][d] + outb[3][m][d];
        out[((size_t)(b * SEQ + q0 + m)) * PDIM + head * HD + d] = (_Float16)s;
    }
}

// ---------------------------------------------------------------------------
// Gather last 512 tokens per batch into contiguous (2048, 384) f16
// ---------------------------------------------------------------------------
__global__ void gather_last(const _Float16* __restrict__ X, _Float16* __restrict__ G) {
    int row = blockIdx.x;                 // 0..2047
    int b = row >> 9, s = K_ENC + (row & 511);
    const _Float16* src = X + ((size_t)(b * SEQ + s)) * PDIM;
    _Float16* dst = G + (size_t)row * PDIM;
    for (int c = threadIdx.x; c < PDIM; c += blockDim.x) dst[c] = src[c];
}

// ---------------------------------------------------------------------------
static inline void launch_gemm(const _Float16* A, const _Float16* Wt,
                               const float* bias, const float* R, float* O32, _Float16* O16,
                               int M, int N, int K, int fuse, hipStream_t s) {
    dim3 grid(N / 64, M / 128);
    gemm_wmma_kernel<<<grid, 128, 0, s>>>(A, Wt, bias, R, O32, O16, M, N, K, fuse);
}

static inline void launch_cvt_t(const float* src, _Float16* dst, int K, int N, int L,
                                hipStream_t s) {
    long total = (long)L * K * N;
    cvt_transpose<<<(int)((total + 255) / 256), 256, 0, s>>>(src, dst, K, N, total);
}

extern "C" void kernel_launch(void* const* d_in, const int* in_sizes, int n_in,
                              void* d_out, int out_size, void* d_ws, size_t ws_size,
                              hipStream_t stream) {
    (void)in_sizes; (void)n_in; (void)out_size; (void)ws_size;
    const float* x       = (const float*)d_in[0];
    const float* pe      = (const float*)d_in[1];
    const int*   me      = (const int*)  d_in[2];
    const int*   mp      = (const int*)  d_in[3];
    const float* W_embed = (const float*)d_in[4];
    const float* b_embed = (const float*)d_in[5];
    const float* mtok    = (const float*)d_in[6];
    const float* ln1_w   = (const float*)d_in[7];
    const float* ln1_b   = (const float*)d_in[8];
    const float* qkv_w   = (const float*)d_in[9];
    const float* qkv_b   = (const float*)d_in[10];
    const float* proj_w  = (const float*)d_in[11];
    const float* proj_b  = (const float*)d_in[12];
    const float* ln2_w   = (const float*)d_in[13];
    const float* ln2_b   = (const float*)d_in[14];
    const float* fc1_w   = (const float*)d_in[15];
    const float* fc1_b   = (const float*)d_in[16];
    const float* fc2_w   = (const float*)d_in[17];
    const float* fc2_b   = (const float*)d_in[18];
    const float* lnf_w   = (const float*)d_in[19];
    const float* lnf_b   = (const float*)d_in[20];
    const float* W_out   = (const float*)d_in[21];
    const float* b_out   = (const float*)d_in[22];

    char* ws = (char*)d_ws;
    size_t off = 0;
    auto alloc = [&](size_t bytes) -> void* {
        void* p = (void*)(ws + off);
        off += (bytes + 255) & ~(size_t)255;
        return p;
    };

    _Float16* X0h   = (_Float16*)alloc((size_t)ROWS * EMBED * 2);   // pre-embed tokens f16
    float*    xb    = (float*)   alloc((size_t)ROWS * PDIM * 4);    // residual stream (fp32)
    float*    x2    = (float*)   alloc((size_t)ROWS * PDIM * 4);    // residual stream (fp32)
    _Float16* bufAh = (_Float16*)alloc((size_t)ROWS * PDIM * 2);    // LN output f16
    _Float16* abufh = (_Float16*)alloc((size_t)ROWS * PDIM * 2);    // attention output f16
    _Float16* qkvh  = (_Float16*)alloc((size_t)ROWS * H3 * 2);
    _Float16* vTh   = (_Float16*)alloc((size_t)BB * HEADS * HD * SEQ * 2);
    _Float16* hbuf  = (_Float16*)alloc((size_t)ROWS * HID * 2);     // MLP hidden f16
    _Float16* gbufh = (_Float16*)alloc((size_t)2048 * PDIM * 2);
    _Float16* hWemb = (_Float16*)alloc((size_t)EMBED * PDIM * 2);           // transposed
    _Float16* hqkv  = (_Float16*)alloc((size_t)DEPTH * PDIM * H3 * 2);      // transposed
    _Float16* hproj = (_Float16*)alloc((size_t)DEPTH * PDIM * PDIM * 2);    // transposed
    _Float16* hfc1  = (_Float16*)alloc((size_t)DEPTH * PDIM * HID * 2);     // transposed
    _Float16* hfc2  = (_Float16*)alloc((size_t)DEPTH * HID * PDIM * 2);     // transposed
    _Float16* hWout = (_Float16*)alloc((size_t)PDIM * EMBED * 2);           // transposed

    // weights -> f16 transposed (N,K)
    launch_cvt_t(W_embed, hWemb, EMBED, PDIM, 1, stream);
    launch_cvt_t(qkv_w,   hqkv,  PDIM, H3,  DEPTH, stream);
    launch_cvt_t(proj_w,  hproj, PDIM, PDIM, DEPTH, stream);
    launch_cvt_t(fc1_w,   hfc1,  PDIM, HID, DEPTH, stream);
    launch_cvt_t(fc2_w,   hfc2,  HID, PDIM, DEPTH, stream);
    launch_cvt_t(W_out,   hWout, PDIM, EMBED, 1, stream);

    // tokens + PE (f16), then embed to PDIM (fp32 residual stream)
    build_x0<<<ROWS, 256, 0, stream>>>(x, pe, me, mp, mtok, X0h);
    launch_gemm(X0h, hWemb, b_embed, nullptr, xb, nullptr, ROWS, PDIM, EMBED, 0, stream);

    for (int i = 0; i < DEPTH; ++i) {
        ln_kernel<<<ROWS, 128, 0, stream>>>(xb, ln1_w + i * PDIM, ln1_b + i * PDIM, bufAh);
        launch_gemm(bufAh, hqkv + (size_t)i * PDIM * H3, qkv_b + i * H3,
                    nullptr, nullptr, qkvh, ROWS, H3, PDIM, 0, stream);
        transpose_v<<<(BB * HEADS * HD * SEQ + 255) / 256, 256, 0, stream>>>(qkvh, vTh);
        attn_kernel<<<dim3(SEQ / 16, HEADS, BB), 128, 0, stream>>>(qkvh, vTh, abufh);
        launch_gemm(abufh, hproj + (size_t)i * PDIM * PDIM, proj_b + i * PDIM,
                    xb, x2, nullptr, ROWS, PDIM, PDIM, 2, stream);
        ln_kernel<<<ROWS, 128, 0, stream>>>(x2, ln2_w + i * PDIM, ln2_b + i * PDIM, bufAh);
        launch_gemm(bufAh, hfc1 + (size_t)i * PDIM * HID, fc1_b + i * HID,
                    nullptr, nullptr, hbuf, ROWS, HID, PDIM, 1, stream);
        launch_gemm(hbuf, hfc2 + (size_t)i * HID * PDIM, fc2_b + i * PDIM,
                    x2, xb, nullptr, ROWS, PDIM, HID, 2, stream);
    }

    ln_kernel<<<ROWS, 128, 0, stream>>>(xb, lnf_w, lnf_b, bufAh);
    gather_last<<<2048, 128, 0, stream>>>(bufAh, gbufh);
    launch_gemm(gbufh, hWout, b_out, nullptr, (float*)d_out, nullptr,
                2048, EMBED, PDIM, 0, stream);
}
